// MIL_Cluster_FC_47519518163083
// MI455X (gfx1250) — matrix operand order repaired and checked by reading, not compiled, not run on hardware
//
#include <hip/hip_runtime.h>
#include <math.h>

#define N_TOK 20000
#define DIM   1024
#define HID   512
#define DATTN 256
#define NCLU  10
#define NCLSZ 4
#define TILE_M 64

// padded LDS strides (in uints) chosen so rows stay 16B-aligned AND
// lane strides hit 16 distinct banks: 20*l mod 64 / 4*r mod 64 are distinct
#define WSTR  20            // weight slab row stride (uints)
#define ASTR  20            // A slab row stride (uints)
#define H1STR 260           // h1 row stride (uints) = 520 halves

typedef __attribute__((ext_vector_type(16))) __bf16       v16bf;
typedef __attribute__((ext_vector_type(8)))  float        v8f;
typedef __attribute__((ext_vector_type(8)))  unsigned int v8u;

// ---- f32 -> bf16 (round to nearest even), packed pair helpers -------------
__device__ __forceinline__ unsigned short f2bf(float f) {
    unsigned u = __float_as_uint(f);
    unsigned r = u + 0x7FFFu + ((u >> 16) & 1u);
    return (unsigned short)(r >> 16);
}
__device__ __forceinline__ unsigned packbf(float a, float b) {
    return (unsigned)f2bf(a) | ((unsigned)f2bf(b) << 16);
}

// ---------------------------------------------------------------------------
// Workspace bucketing kernels
// ---------------------------------------------------------------------------
__global__ void k_zero(float* __restrict__ h_sum, int* __restrict__ counts) {
    int i = blockIdx.x * blockDim.x + threadIdx.x;
    if (i < NCLU * HID) h_sum[i] = 0.0f;
    if (i < NCLU)       counts[i] = 0;
}

__global__ void k_count(const int* __restrict__ cid, int* __restrict__ counts) {
    int i = blockIdx.x * blockDim.x + threadIdx.x;
    if (i < N_TOK) atomicAdd(&counts[cid[i]], 1);
}

__global__ void k_prefix(const int* __restrict__ counts,
                         int* __restrict__ offsets, int* __restrict__ cursor) {
    if (threadIdx.x == 0) {
        int s = 0;
        for (int c = 0; c < NCLU; ++c) { offsets[c] = s; cursor[c] = s; s += counts[c]; }
        offsets[NCLU] = s;
    }
}

__global__ void k_scatter(const int* __restrict__ cid,
                          int* __restrict__ cursor, int* __restrict__ order) {
    int i = blockIdx.x * blockDim.x + threadIdx.x;
    if (i < N_TOK) {
        int c = cid[i];
        int p = atomicAdd(&cursor[c], 1);
        order[p] = i;
    }
}

// ---------------------------------------------------------------------------
// Expert GEMM kernel: per-cluster 64-token tiles through W1/W2, bf16 WMMA.
// 8 waves: rt = wv&3 (16-row tile), g = wv>>2 (256-wide column half).
// Each wave owns 16 accumulator fragments (128 VGPRs).
// ---------------------------------------------------------------------------
__device__ __forceinline__ void stage_w(const float* __restrict__ Wg, int k0,
                                        unsigned* __restrict__ w_lds, int tid) {
    // 16 k-pairs x 128 float4 column-chunks, coalesced over n
    for (int it = tid; it < 16 * 128; it += 256) {
        int i  = it >> 7;                 // k-pair index
        int n4 = (it & 127) << 2;         // column base
        const float4 f0 = *(const float4*)(Wg + (size_t)(k0 + 2 * i) * HID + n4);
        const float4 f1 = *(const float4*)(Wg + (size_t)(k0 + 2 * i + 1) * HID + n4);
        w_lds[(n4 + 0) * WSTR + i] = packbf(f0.x, f1.x);
        w_lds[(n4 + 1) * WSTR + i] = packbf(f0.y, f1.y);
        w_lds[(n4 + 2) * WSTR + i] = packbf(f0.z, f1.z);
        w_lds[(n4 + 3) * WSTR + i] = packbf(f0.w, f1.w);
    }
}

// One K=32 step: A fragment (2x ds_load_b128), 16 WMMAs against staged B slab.
__device__ __forceinline__ void wmma_step(const unsigned* __restrict__ Arow,
                                          const unsigned* __restrict__ w_lds,
                                          int g, int l15, int half, v8f* acc) {
    uint4 a0 = *(const uint4*)(Arow + half * 8);
    uint4 a1 = *(const uint4*)(Arow + half * 8 + 4);
    v8u af;
    af[0] = a0.x; af[1] = a0.y; af[2] = a0.z; af[3] = a0.w;
    af[4] = a1.x; af[5] = a1.y; af[6] = a1.z; af[7] = a1.w;
    v16bf A = __builtin_bit_cast(v16bf, af);
#pragma unroll
    for (int nt = 0; nt < 16; ++nt) {
        const unsigned* Brow = &w_lds[(g * 256 + nt * 16 + l15) * WSTR + half * 8];
        uint4 b0 = *(const uint4*)(Brow);
        uint4 b1 = *(const uint4*)(Brow + 4);
        v8u bu;
        bu[0] = b0.x; bu[1] = b0.y; bu[2] = b0.z; bu[3] = b0.w;
        bu[4] = b1.x; bu[5] = b1.y; bu[6] = b1.z; bu[7] = b1.w;
        v16bf B = __builtin_bit_cast(v16bf, bu);
        acc[nt] = __builtin_amdgcn_wmma_f32_16x16x32_bf16(
            false, A, false, B, (short)0, acc[nt], false, false);
    }
}

__global__ __launch_bounds__(256)
void k_expert(const float* __restrict__ x,
              const float* __restrict__ W1, const float* __restrict__ b1,
              const float* __restrict__ W2, const float* __restrict__ b2,
              const int* __restrict__ counts, const int* __restrict__ offsets,
              const int* __restrict__ order, float* __restrict__ h_sum) {
    __shared__ __align__(16) unsigned       w_lds[HID * WSTR];        // 40 KB
    __shared__ __align__(16) unsigned short h1_lds[TILE_M * 2 * H1STR]; // 65 KB
    __shared__ __align__(16) unsigned       a_lds[TILE_M * ASTR];     // 5 KB
    __shared__ int                          tok[TILE_M];

    const int c  = blockIdx.y;
    const int nc = counts[c];
    const int m0 = blockIdx.x * TILE_M;
    if (m0 >= nc) return;                               // uniform early exit
    const int offs = offsets[c];
    const int tid  = threadIdx.x;
    const int lane = tid & 31, wv = tid >> 5;
    const int rt = wv & 3;                              // 16-row tile (0..3)
    const int g  = wv >> 2;                             // 256-wide column half
    const int half = lane >> 4, l15 = lane & 15;

    if (tid < TILE_M) tok[tid] = (m0 + tid < nc) ? order[offs + m0 + tid] : -1;
    __syncthreads();

    v8f acc[16];
    const v8f vzero = {};
#pragma unroll
    for (int t = 0; t < 16; ++t) acc[t] = vzero;

    // ---------------- GEMM1: h1 = relu(x @ W1[c] + b1[c]) ----------------
    const float* W1c = W1 + (size_t)c * DIM * HID;
    for (int k0 = 0; k0 < DIM; k0 += 32) {
        {   // stage A: one thread per (row, 4-k-pair group); b128 in, b128 out
            int row = tid >> 2;
            int i4  = tid & 3;
            int t = tok[row];
            uint4 pk = make_uint4(0u, 0u, 0u, 0u);
            if (t >= 0) {
                const float* xr = x + (size_t)t * DIM + k0 + 8 * i4;
                float4 a = *(const float4*)(xr);
                float4 b = *(const float4*)(xr + 4);
                pk = make_uint4(packbf(a.x, a.y), packbf(a.z, a.w),
                                packbf(b.x, b.y), packbf(b.z, b.w));
            }
            *(uint4*)&a_lds[row * ASTR + 4 * i4] = pk;
        }
        stage_w(W1c, k0, w_lds, tid);                           // stage B
        if (k0 + 32 < DIM) {                                    // prefetch next slab
            const char* nxt = (const char*)(W1c + (size_t)(k0 + 32) * HID);
            for (int p = tid; p < 1024; p += 256)
                __builtin_prefetch(nxt + p * 64, 0, 0);
        }
        __syncthreads();
        wmma_step(&a_lds[(rt * 16 + l15) * ASTR], w_lds, g, l15, half, acc);
        __syncthreads();
    }
    // bias + relu -> bf16 h1 in LDS
#pragma unroll
    for (int nt = 0; nt < 16; ++nt) {
        int n = g * 256 + nt * 16 + l15;
        float bv = b1[c * HID + n];
#pragma unroll
        for (int j = 0; j < 8; ++j) {
            int m = rt * 16 + half * 8 + j;                     // C-frag row map
            float v = acc[nt][j] + bv;
            h1_lds[m * (2 * H1STR) + n] = f2bf(v > 0.0f ? v : 0.0f);
        }
        acc[nt] = vzero;
    }
    __syncthreads();

    // ---------------- GEMM2: h2 = relu(h1 @ W2[c] + b2[c]) ----------------
    const float*    W2c = W2 + (size_t)c * HID * HID;
    const unsigned* h1u = (const unsigned*)h1_lds;
    for (int ks = 0; ks < HID / 32; ++ks) {
        stage_w(W2c, ks * 32, w_lds, tid);
        if (ks * 32 + 32 < HID) {
            const char* nxt = (const char*)(W2c + (size_t)(ks * 32 + 32) * HID);
            for (int p = tid; p < 1024; p += 256)
                __builtin_prefetch(nxt + p * 64, 0, 0);
        }
        __syncthreads();
        wmma_step(h1u + (size_t)(rt * 16 + l15) * H1STR + ks * 16,
                  w_lds, g, l15, half, acc);
        __syncthreads();
    }
    // bias + relu + masked per-column partial sums -> atomic accumulate
#pragma unroll
    for (int nt = 0; nt < 16; ++nt) {
        int n = g * 256 + nt * 16 + l15;
        float bv = b2[c * HID + n];
        float s = 0.0f;
#pragma unroll
        for (int j = 0; j < 8; ++j) {
            int m = rt * 16 + half * 8 + j;
            if (m0 + m < nc) {
                float v = acc[nt][j] + bv;
                s += (v > 0.0f ? v : 0.0f);
            }
        }
        atomicAdd(&h_sum[c * HID + n], s);
    }
}

// ---------------------------------------------------------------------------
// Head: mean-pool, fc+gated attention, softmax over clusters, rho, classifier
// ---------------------------------------------------------------------------
__global__ __launch_bounds__(256)
void k_head(const int* __restrict__ counts, const float* __restrict__ h_sum,
            const float* __restrict__ fc_W, const float* __restrict__ fc_b,
            const float* __restrict__ Wa, const float* __restrict__ ba,
            const float* __restrict__ Wb, const float* __restrict__ bbv,
            const float* __restrict__ Wc, const float* __restrict__ bc,
            const float* __restrict__ rho_W, const float* __restrict__ rho_b,
            const float* __restrict__ cls_W, const float* __restrict__ cls_b,
            float* __restrict__ out) {
    __shared__ float hc[NCLU * HID];
    __shared__ float hh[NCLU * HID];
    __shared__ float att[NCLU * DATTN];
    __shared__ float wA[NCLU];
    __shared__ float hp[HID];
    __shared__ float hr[DATTN];
    const int tid = threadIdx.x;

    for (int i = tid; i < NCLU * HID; i += 256) {
        int c = i / HID;
        float cnt = (float)counts[c]; if (cnt < 1.0f) cnt = 1.0f;
        hc[i] = h_sum[i] / cnt;
    }
    __syncthreads();
    for (int i = tid; i < NCLU * HID; i += 256) {               // h = relu(fc)
        int c = i / HID, h = i % HID;
        float s = fc_b[h];
        for (int k = 0; k < HID; ++k) s += hc[c * HID + k] * fc_W[k * HID + h];
        hh[i] = s > 0.0f ? s : 0.0f;
    }
    __syncthreads();
    for (int i = tid; i < NCLU * DATTN; i += 256) {             // gated attention
        int c = i / DATTN, d = i % DATTN;
        float sa = ba[d], sb = bbv[d];
        for (int k = 0; k < HID; ++k) {
            float hv = hh[c * HID + k];
            sa += hv * Wa[k * DATTN + d];
            sb += hv * Wb[k * DATTN + d];
        }
        att[i] = tanhf(sa) * (1.0f / (1.0f + expf(-sb)));
    }
    __syncthreads();
    if (tid < NCLU) {
        float s = bc[0];
        for (int d = 0; d < DATTN; ++d) s += att[tid * DATTN + d] * Wc[d];
        wA[tid] = s;
    }
    __syncthreads();
    if (tid == 0) {                                             // softmax over C
        float mx = wA[0];
        for (int c = 1; c < NCLU; ++c) mx = fmaxf(mx, wA[c]);
        float den = 0.0f;
        for (int c = 0; c < NCLU; ++c) { wA[c] = expf(wA[c] - mx); den += wA[c]; }
        for (int c = 0; c < NCLU; ++c) wA[c] /= den;
    }
    __syncthreads();
    for (int h = tid; h < HID; h += 256) {                      // h_path = A @ h
        float s = 0.0f;
        for (int c = 0; c < NCLU; ++c) s += wA[c] * hh[c * HID + h];
        hp[h] = s;
    }
    __syncthreads();
    for (int d = tid; d < DATTN; d += 256) {                    // hr = relu(rho)
        float s = rho_b[d];
        for (int k = 0; k < HID; ++k) s += hp[k] * rho_W[k * DATTN + d];
        hr[d] = s > 0.0f ? s : 0.0f;
    }
    __syncthreads();
    if (tid == 0) {                                             // classifier
        float lg[NCLSZ];
        for (int j = 0; j < NCLSZ; ++j) {
            float s = cls_b[j];
            for (int d = 0; d < DATTN; ++d) s += hr[d] * cls_W[d * NCLSZ + j];
            lg[j] = s;
        }
        float mx = lg[0]; int am = 0;
        for (int j = 1; j < NCLSZ; ++j) if (lg[j] > mx) { mx = lg[j]; am = j; }
        float den = 0.0f, pr[NCLSZ];
        for (int j = 0; j < NCLSZ; ++j) { pr[j] = expf(lg[j] - mx); den += pr[j]; }
        for (int j = 0; j < NCLSZ; ++j) { out[j] = lg[j]; out[NCLSZ + j] = pr[j] / den; }
        out[2 * NCLSZ] = (float)am;
    }
}

// ---------------------------------------------------------------------------
extern "C" void kernel_launch(void* const* d_in, const int* in_sizes, int n_in,
                              void* d_out, int out_size, void* d_ws, size_t ws_size,
                              hipStream_t stream) {
    const float* x    = (const float*)d_in[0];
    const int*   cid  = (const int*)  d_in[1];
    const float* W1   = (const float*)d_in[2];
    const float* b1   = (const float*)d_in[3];
    const float* W2   = (const float*)d_in[4];
    const float* b2   = (const float*)d_in[5];
    const float* fcW  = (const float*)d_in[6];
    const float* fcb  = (const float*)d_in[7];
    const float* Wa   = (const float*)d_in[8];
    const float* ba   = (const float*)d_in[9];
    const float* Wb   = (const float*)d_in[10];
    const float* bb   = (const float*)d_in[11];
    const float* Wc   = (const float*)d_in[12];
    const float* bc   = (const float*)d_in[13];
    const float* rhoW = (const float*)d_in[14];
    const float* rhob = (const float*)d_in[15];
    const float* clsW = (const float*)d_in[16];
    const float* clsb = (const float*)d_in[17];

    char*  ws      = (char*)d_ws;
    int*   counts  = (int*)(ws + 0);
    int*   offsets = (int*)(ws + 64);
    int*   cursor  = (int*)(ws + 128);
    float* h_sum   = (float*)(ws + 256);
    int*   order   = (int*)(ws + 256 + NCLU * HID * sizeof(float));

    k_zero<<<dim3((NCLU * HID + 255) / 256), 256, 0, stream>>>(h_sum, counts);
    k_count<<<dim3((N_TOK + 255) / 256), 256, 0, stream>>>(cid, counts);
    k_prefix<<<1, 32, 0, stream>>>(counts, offsets, cursor);
    k_scatter<<<dim3((N_TOK + 255) / 256), 256, 0, stream>>>(cid, cursor, order);

    dim3 grid((N_TOK + TILE_M - 1) / TILE_M, NCLU);
    k_expert<<<grid, 256, 0, stream>>>(x, W1, b1, W2, b2,
                                       counts, offsets, order, h_sum);
    k_head<<<1, 256, 0, stream>>>(counts, h_sum, fcW, fcb, Wa, ba, Wb, bb,
                                  Wc, bc, rhoW, rhob, clsW, clsb, (float*)d_out);
}
